// InformationRouter_24661702214141
// MI455X (gfx1250) — matrix operand order repaired and checked by reading, not compiled, not run on hardware
//
#include <hip/hip_runtime.h>
#include <hip/hip_bf16.h>
#include <math.h>

typedef __attribute__((ext_vector_type(16))) _Float16 v16h;
typedef __attribute__((ext_vector_type(4)))  _Float16 v4h;
typedef __attribute__((ext_vector_type(8)))  float    v8f;

#define BATCH  8
#define HEADS  8
#define SEQ    512
#define DIM    256
#define BH     (BATCH*HEADS)        // 64
#define TOKENS (BH*SEQ)             // 32768
#define THRV   0.7f

// ---------------------------------------------------------------------------
// Fragment loaders (CDNA5 wave32 WMMA 16x16x32 f16 layouts, ISA 05_wmma §7.12.2)
//
// A (16x32, MxK), f16: lane l (g=l>>4, m=l&15) holds row m,
//   Ks {16g+0..15} across the 8 VGPRs' halves (same k order as B).
//   => one contiguous 16-float run from the row-major f32 source, cvt to f16.
// B (32x16, KxN), f16: lane l holds column n=l&15, Ks {16g+0..15} across the
//   8 VGPRs' halves => 16 contiguous f16 from the (N,K) row-major weight.
// C/D (16x16 f32): lane l, VGPR r -> M = r + 8*(l>>4), N = l&15.
// ---------------------------------------------------------------------------
__device__ __forceinline__ v16h load_a_frag(const float* __restrict__ rowptr,
                                            int kb, int lane) {
  int g = (lane >> 4) & 1;
  const float* p0 = rowptr + kb + 16 * g;
  float4 f0 = *(const float4*)(p0);
  float4 f1 = *(const float4*)(p0 + 4);
  float4 f2 = *(const float4*)(p0 + 8);
  float4 f3 = *(const float4*)(p0 + 12);
  v16h a;
  a[0] = (_Float16)f0.x;  a[1] = (_Float16)f0.y;  a[2] = (_Float16)f0.z;  a[3] = (_Float16)f0.w;
  a[4] = (_Float16)f1.x;  a[5] = (_Float16)f1.y;  a[6] = (_Float16)f1.z;  a[7] = (_Float16)f1.w;
  a[8] = (_Float16)f2.x;  a[9] = (_Float16)f2.y;  a[10] = (_Float16)f2.z; a[11] = (_Float16)f2.w;
  a[12] = (_Float16)f3.x; a[13] = (_Float16)f3.y; a[14] = (_Float16)f3.z; a[15] = (_Float16)f3.w;
  return a;
}

__device__ __forceinline__ v16h load_b_frag(const _Float16* __restrict__ W,
                                            int ldk, int nb, int kb, int lane) {
  int g = (lane >> 4) & 1;
  const _Float16* p = W + (size_t)(nb + (lane & 15)) * ldk + kb + 16 * g;
  return *(const v16h*)p;   // 32B contiguous -> 2x ds_load_b128
}

// ---------------------------------------------------------------------------
// Kernel 1/3: route scores.  scores[tok] = relu(x @ W1^T + b1) @ w2 + b2
// W1 (512x256) staged to LDS as f16 (256KB, fits CDNA5's 320KB WGP LDS).
// Each wave owns one 16-token M-tile; 32 N-tiles x 8 K-steps of WMMA, fused
// relu+dot epilogue, 16-lane shfl_xor reduction to per-token scalar.
// ---------------------------------------------------------------------------
__global__ __launch_bounds__(256) void score_kernel(
    const float* __restrict__ X,      // (TOKENS, 256)
    const float* __restrict__ w1,     // (512, 256)
    const float* __restrict__ b1,     // (512)
    const float* __restrict__ w2,     // (1, 512)
    const float* __restrict__ b2,     // (1)
    float* __restrict__ scores)       // (TOKENS)
{
  extern __shared__ float4 smem4[];
  _Float16* w1h = (_Float16*)smem4;
  int tid = threadIdx.x;

  const float4* src = (const float4*)w1;
  v4h* dst = (v4h*)w1h;
  for (int i = tid; i < 512 * 256 / 4; i += 256) {
    float4 f = src[i];
    v4h h; h[0] = (_Float16)f.x; h[1] = (_Float16)f.y; h[2] = (_Float16)f.z; h[3] = (_Float16)f.w;
    dst[i] = h;
  }
  __syncthreads();

  int wave = tid >> 5, lane = tid & 31;
  int g = lane >> 4;
  float b2v = b2[0];
  const int ntiles = TOKENS / 16;   // 2048 (16-token tiles never cross a (b,h))

  for (int tile = blockIdx.x * 8 + wave; tile < ntiles; tile += gridDim.x * 8) {
    const float* rowptr = X + (size_t)(tile * 16 + (lane & 15)) * DIM;
    v16h afr[8];
#pragma unroll
    for (int ks = 0; ks < 8; ++ks) afr[ks] = load_a_frag(rowptr, ks * 32, lane);

    v8f psum = {};
#pragma unroll 4
    for (int nt = 0; nt < 32; ++nt) {
      v8f c = {};
#pragma unroll
      for (int ks = 0; ks < 8; ++ks) {
        v16h b = load_b_frag(w1h, 256, nt * 16, ks * 32, lane);
        c = __builtin_amdgcn_wmma_f32_16x16x32_f16(false, afr[ks], false, b,
                                                   (short)0, c, false, false);
      }
      int n = nt * 16 + (lane & 15);
      float bias = b1[n];
      float wv = w2[n];
#pragma unroll
      for (int r = 0; r < 8; ++r) {
        float hv = c[r] + bias;
        hv = hv > 0.f ? hv : 0.f;
        psum[r] += hv * wv;
      }
    }
#pragma unroll
    for (int r = 0; r < 8; ++r) {
      float v = psum[r];
      v += __shfl_xor(v, 1);
      v += __shfl_xor(v, 2);
      v += __shfl_xor(v, 4);
      v += __shfl_xor(v, 8);
      psum[r] = v;
    }
    if ((lane & 15) == 0) {
#pragma unroll
      for (int r = 0; r < 8; ++r)
        scores[tile * 16 + r + 8 * g] = psum[r] + b2v;
    }
  }
}

// ---------------------------------------------------------------------------
// Kernel 2: fused bottleneck MLP + flow + blend -> optimized.
// A = concat(states[t], states[t+1]) (K=512), B = wb1 (256x512) in LDS f16.
// 16 N-tile accumulators stay in VGPRs (128), K outer so A frags load once.
// Epilogue: sigmoid -> impact gate (uses gradient of routing_scores) ->
// per-row blend written straight to `optimized`; tile j==0 also copies row 0.
// ---------------------------------------------------------------------------
__global__ __launch_bounds__(256) void opt_kernel(
    const float* __restrict__ states,
    const float* __restrict__ rscores,  // (BH, 512)
    const float* __restrict__ wb1,      // (256, 512)
    const float* __restrict__ bb1,      // (256)
    const float* __restrict__ wb2,      // (1, 256)
    const float* __restrict__ bb2,      // (1)
    float* __restrict__ optimized)      // (TOKENS, 256)
{
  extern __shared__ float4 smem4[];
  _Float16* wh = (_Float16*)smem4;
  int tid = threadIdx.x;

  const float4* src = (const float4*)wb1;
  v4h* dst = (v4h*)wh;
  for (int i = tid; i < 256 * 512 / 4; i += 256) {
    float4 f = src[i];
    v4h h; h[0] = (_Float16)f.x; h[1] = (_Float16)f.y; h[2] = (_Float16)f.z; h[3] = (_Float16)f.w;
    dst[i] = h;
  }
  __syncthreads();

  int wave = tid >> 5, lane = tid & 31;
  int g = lane >> 4, lm = lane & 15;
  float bb2v = bb2[0];
  const int ntilesTotal = BH * 32;   // 2048 tiles of 16 pair-indices t

  for (int tile = blockIdx.x * 8 + wave; tile < ntilesTotal; tile += gridDim.x * 8) {
    int bh = tile >> 5, j = tile & 31;
    int t_lane = j * 16 + lm;
    int tc = t_lane > 510 ? 510 : t_lane;          // clamp the one invalid row
    const float* r1 = states + (size_t)(bh * SEQ + tc) * DIM;
    const float* r2 = r1 + DIM;

    v8f acc[16];
#pragma unroll
    for (int nt = 0; nt < 16; ++nt) acc[nt] = (v8f){};
#pragma unroll
    for (int ks = 0; ks < 16; ++ks) {
      int kb = ks * 32;
      v16h a = (kb < 256) ? load_a_frag(r1, kb, lane)
                          : load_a_frag(r2, kb - 256, lane);
#pragma unroll
      for (int nt = 0; nt < 16; ++nt) {
        v16h b = load_b_frag(wh, 512, nt * 16, kb, lane);
        acc[nt] = __builtin_amdgcn_wmma_f32_16x16x32_f16(false, a, false, b,
                                                         (short)0, acc[nt], false, false);
      }
    }

    v8f psum = {};
#pragma unroll
    for (int nt = 0; nt < 16; ++nt) {
      int n = nt * 16 + lm;
      float bias = bb1[n];
      float wv = wb2[n];
#pragma unroll
      for (int r = 0; r < 8; ++r) {
        float hv = acc[nt][r] + bias;
        hv = hv > 0.f ? hv : 0.f;
        psum[r] += hv * wv;
      }
    }
#pragma unroll
    for (int r = 0; r < 8; ++r) {
      float v = psum[r];
      v += __shfl_xor(v, 1);
      v += __shfl_xor(v, 2);
      v += __shfl_xor(v, 4);
      v += __shfl_xor(v, 8);
      psum[r] = v;   // all 16 lanes of the group now hold the full sum
    }

    const float* scr = rscores + (size_t)bh * SEQ;
#pragma unroll
    for (int r = 0; r < 8; ++r) {
      int m = r + 8 * g;
      int t = j * 16 + m;
      if (t <= 510) {
        float pre = psum[r] + bb2v;
        float bn = 1.f / (1.f + expf(-pre));
        // jnp.gradient along seq, then [:-1]: one-sided at t==0, central else
        float fl = (t == 0) ? (scr[1] - scr[0]) : 0.5f * (scr[t + 1] - scr[t - 1]);
        float imp = (bn > THRV) ? fl : 0.f;
        float aa = fabsf(imp);
        const float* s1 = states + (size_t)(bh * SEQ + t) * DIM;
        const float* s2 = s1 + DIM;
        float* op = optimized + (size_t)(bh * SEQ + t + 1) * DIM;
        int c0 = lm * 16;                     // 16 lanes x 16 cols = 256
        if (aa > THRV) {
#pragma unroll
          for (int q = 0; q < 4; ++q) {
            float4 x1 = *(const float4*)(s1 + c0 + 4 * q);
            float4 x2 = *(const float4*)(s2 + c0 + 4 * q);
            float4 res;
            res.x = (1.f - aa) * x2.x + aa * x1.x;
            res.y = (1.f - aa) * x2.y + aa * x1.y;
            res.z = (1.f - aa) * x2.z + aa * x1.z;
            res.w = (1.f - aa) * x2.w + aa * x1.w;
            *(float4*)(op + c0 + 4 * q) = res;
          }
        } else {
#pragma unroll
          for (int q = 0; q < 4; ++q)
            *(float4*)(op + c0 + 4 * q) = *(const float4*)(s2 + c0 + 4 * q);
        }
      }
    }
    if (j == 0 && g == 0) {   // optimized row 0 = states row 0
      const float* s0 = states + (size_t)bh * SEQ * DIM;
      float* op = optimized + (size_t)bh * SEQ * DIM;
      int c0 = lm * 16;
#pragma unroll
      for (int q = 0; q < 4; ++q)
        *(float4*)(op + c0 + 4 * q) = *(const float4*)(s0 + c0 + 4 * q);
    }
  }
}

// ---------------------------------------------------------------------------
// Kernel 4: path. The scan's GRU carry is dead code; path[0]=0 and
// path[1..] = successive argmax of scores_all with picked indices masked
// (= selection sort descending, first-index tie-break like jnp.argmax).
// One wave per (b,h): 512 scores in 16 regs/lane, shfl_xor argmax butterfly.
// ---------------------------------------------------------------------------
__global__ __launch_bounds__(32) void path_kernel(
    const float* __restrict__ scores_all,   // (BH, 512)
    float* __restrict__ path)               // (BH, 512), indices as float
{
  int bh = blockIdx.x;
  int lane = threadIdx.x;
  const float* s = scores_all + (size_t)bh * SEQ;
  float v[16];
#pragma unroll
  for (int k = 0; k < 16; ++k) v[k] = s[lane + 32 * k];
  if (lane == 0) {
    v[0] = -__FLT_MAX__;          // index 0 pre-masked
    path[(size_t)bh * SEQ] = 0.0f;
  }
  for (int step = 1; step < SEQ; ++step) {
    float bv = v[0];
    int bi = lane;
#pragma unroll
    for (int k = 1; k < 16; ++k) {
      int idx = lane + 32 * k;
      if (v[k] > bv) { bv = v[k]; bi = idx; }   // ascending k => first-idx wins ties
    }
#pragma unroll
    for (int off = 16; off > 0; off >>= 1) {
      float ov = __shfl_xor(bv, off);
      int   oi = __shfl_xor(bi, off);
      if (ov > bv || (ov == bv && oi < bi)) { bv = ov; bi = oi; }
    }
    if (lane == 0) path[(size_t)bh * SEQ + step] = (float)bi;
#pragma unroll
    for (int k = 0; k < 16; ++k)
      if (bi == lane + 32 * k) v[k] = -__FLT_MAX__;
  }
}

// ---------------------------------------------------------------------------
extern "C" void kernel_launch(void* const* d_in, const int* in_sizes, int n_in,
                              void* d_out, int out_size, void* d_ws, size_t ws_size,
                              hipStream_t stream) {
  (void)in_sizes; (void)n_in; (void)out_size; (void)ws_size;
  const float* states = (const float*)d_in[0];
  const float* w1  = (const float*)d_in[1];
  const float* b1  = (const float*)d_in[2];
  const float* w2  = (const float*)d_in[3];
  const float* b2  = (const float*)d_in[4];
  const float* wb1 = (const float*)d_in[5];
  const float* bb1 = (const float*)d_in[6];
  const float* wb2 = (const float*)d_in[7];
  const float* bb2 = (const float*)d_in[8];
  // d_in[9..16] (GRU weights) are dead w.r.t. the reference's outputs.

  float* out       = (float*)d_out;
  float* optimized = out;                          // (8,8,512,256)
  float* path      = out + (size_t)TOKENS * DIM;   // (8,8,512)
  float* rscores   = path + TOKENS;                // (8,8,512)
  float* sall      = (float*)d_ws;                 // scratch: (8,8,512) f32

  size_t ldsBytes = (size_t)512 * 256 * sizeof(_Float16);  // 256 KB (WGP LDS = 320 KB)

  score_kernel<<<dim3(256), dim3(256), ldsBytes, stream>>>(states, w1, b1, w2, b2, rscores);
  opt_kernel  <<<dim3(256), dim3(256), ldsBytes, stream>>>(states, rscores, wb1, bb1, wb2, bb2, optimized);
  score_kernel<<<dim3(256), dim3(256), ldsBytes, stream>>>(optimized, w1, b1, w2, b2, sall);
  path_kernel <<<dim3(BH),  dim3(32),  0,        stream>>>(sall, path);
}
